// MemoryTransformer_66821101191607
// MI455X (gfx1250) — matrix operand order repaired and checked by reading, not compiled, not run on hardware
//
#include <hip/hip_runtime.h>
#include <cmath>
#include <cstdint>
#include <cstddef>

#define B_    2
#define T_    512
#define D_    768
#define H_    12
#define HD_   64
#define DFF_  3072
#define V_    50257
#define NL_   4
#define MLP_  64
#define NMEM_ 2

typedef __bf16 bf16;
typedef __bf16 bf16x16 __attribute__((ext_vector_type(16)));
typedef float  f32x8   __attribute__((ext_vector_type(8)));

#define GF_BIAS 1
#define GF_GELU 2
#define GF_RES  4

__device__ __forceinline__ f32x8 zero8() {
  f32x8 z;
#pragma unroll
  for (int e = 0; e < 8; ++e) z[e] = 0.0f;
  return z;
}

__device__ __forceinline__ unsigned pack2(float a, float b) {
  union { __bf16 h[2]; unsigned u; } z;
  z.h[0] = (__bf16)a; z.h[1] = (__bf16)b;
  return z.u;
}

// Self-inverse 8-chunk swap inside a 32-element block: [0..7|16..23|8..15|24..31].
// Makes each lane's 16-element A-fragment slice (ka pattern) contiguous.
__device__ __forceinline__ int swz32(int o) {
  int blk = (o >> 3) & 3;
  return (blk == 1) ? (o + 8) : (blk == 2) ? (o - 8) : o;
}

// CDNA5 async global->LDS copy (ISA 15.18.3 op 98, ASYNCcnt).
// lds_addr: wave-relative LDS byte address (low 32 bits of a __shared__ pointer).
__device__ __forceinline__ void async_load_b128(unsigned lds_addr, const void* gaddr) {
  asm volatile("global_load_async_to_lds_b128 %0, %1, off"
               :: "v"(lds_addr), "v"((unsigned long long)(uintptr_t)gaddr)
               : "memory");
}
__device__ __forceinline__ void wait_async0() {
  asm volatile("s_wait_asynccnt 0x0" ::: "memory");
}

// ---- GEMM: C[MxN] = A[MxK] @ B[KxN] (+bias)(+gelu)(+residual) ----
// A: bf16, row-major with swz32 k-permutation (from cvt_a_swz_k).
// B: bf16, tiled column-major (from cvt_b_swz_k): tile (kt,nt) is 64n x 32k,
//    stored at ((nt*Kt + kt)*64 + n)*32 + k, zero-padded in n.
// Block: 128 threads = 4 waves; block tile 64x64; K step 32.
// Tiles staged with GLOBAL_LOAD_ASYNC_TO_LDS_B128; next B tile prefetched.
template <int FLAGS>
__global__ __launch_bounds__(128)
void gemm_bf16(const bf16* __restrict__ A, const bf16* __restrict__ Bt,
               const float* __restrict__ bias, const float* __restrict__ resid,
               float* __restrict__ C, int M, int N, int K, int Kt)
{
  __shared__ __align__(32) bf16 sA[64 * 32];
  __shared__ __align__(32) bf16 sB[64 * 32];
  const int tid  = threadIdx.x;
  const int lane = tid & 31;
  const int wave = tid >> 5;
  const int wm   = (wave >> 1) * 32;
  const int wn   = (wave & 1) * 32;
  const int m0   = blockIdx.y * 64;
  const int n0   = blockIdx.x * 64;
  const int mlo  = lane & 15, half = lane >> 4;

  f32x8 acc[2][2];
#pragma unroll
  for (int i = 0; i < 2; ++i)
#pragma unroll
    for (int j = 0; j < 2; ++j) acc[i][j] = zero8();

  const bf16* btile = Bt + (size_t)blockIdx.x * Kt * 2048;
  const int arow = tid >> 1, ahc = tid & 1;
  const unsigned ldsA0 = (unsigned)(uintptr_t)&sA[arow * 32 + ahc * 16];
  const unsigned ldsA1 = (unsigned)(uintptr_t)&sA[arow * 32 + ahc * 16 + 8];
  const unsigned ldsB0 = (unsigned)(uintptr_t)&sB[tid * 16];
  const unsigned ldsB1 = (unsigned)(uintptr_t)&sB[tid * 16 + 8];

  for (int kb = 0; kb < K; kb += 32, btile += 2048) {
    const bf16* asrc = A + (size_t)(m0 + arow) * K + kb + ahc * 16;
    async_load_b128(ldsA0, asrc);
    async_load_b128(ldsA1, asrc + 8);
    async_load_b128(ldsB0, btile + tid * 16);
    async_load_b128(ldsB1, btile + tid * 16 + 8);
    // hint next B tile into cache while WMMA runs
    __builtin_prefetch(btile + 2048 + tid * 16, 0, 1);
    wait_async0();
    __syncthreads();
    bf16x16 af[2], bfr[2];
    af[0]  = *(const bf16x16*)&sA[(wm + mlo) * 32 + half * 16];
    af[1]  = *(const bf16x16*)&sA[(wm + 16 + mlo) * 32 + half * 16];
    bfr[0] = *(const bf16x16*)&sB[(wn + mlo) * 32 + half * 16];
    bfr[1] = *(const bf16x16*)&sB[(wn + 16 + mlo) * 32 + half * 16];
#pragma unroll
    for (int i = 0; i < 2; ++i)
#pragma unroll
      for (int j = 0; j < 2; ++j)
        acc[i][j] = __builtin_amdgcn_wmma_f32_16x16x32_bf16(
            false, af[i], false, bfr[j], (short)0, acc[i][j], false, false);
    __syncthreads();
  }

#pragma unroll
  for (int i = 0; i < 2; ++i) {
#pragma unroll
    for (int j = 0; j < 2; ++j) {
#pragma unroll
      for (int r = 0; r < 8; ++r) {
        int m = m0 + wm + i * 16 + half * 8 + r;
        int n = n0 + wn + j * 16 + mlo;
        if (n < N) {
          float v = acc[i][j][r];
          if (FLAGS & GF_BIAS) v += bias[n];
          if (FLAGS & GF_GELU) v = 0.5f * v * (1.0f + erff(v * 0.70710678118f));
          size_t idx = (size_t)m * N + n;
          if (FLAGS & GF_RES) v += resid[idx];
          C[idx] = v;
        }
      }
    }
  }
}

// ---- Flash attention: one wave per (16-query tile, head, batch) ----
__global__ __launch_bounds__(32)
void attn_fa(const float* __restrict__ qkv, float* __restrict__ out)
{
  __shared__ __align__(32) bf16 sQ[16 * 64];   // row = query, cols = features (natural order)
  __shared__ __align__(32) bf16 sK[32 * 64];   // key-major (row = key, cols = features)
  __shared__ __align__(32) bf16 sVt[64 * 32];  // transposed: row = dim, cols = keys
  __shared__ __align__(32) bf16 sP[16 * 32];   // row = query, cols = keys (natural order)
  const int lane = threadIdx.x;
  const int qt0  = blockIdx.x * 16;
  const int h    = blockIdx.y;
  const int b    = blockIdx.z;
  const size_t rs = 3 * D_;
  const float* base = qkv + (size_t)(b * T_) * rs + h * HD_;
  const int mlo = lane & 15, half = lane >> 4;

  { // Q tile (16x64) -> bf16 LDS, features in natural order
    int r = lane >> 1;
    int c32 = (lane & 1) * 32;
    const float* src = base + (size_t)(qt0 + r) * rs + c32;
#pragma unroll
    for (int cb = 0; cb < 4; ++cb) {
      int db = cb;
      uint4 p;
      p.x = pack2(src[cb * 8 + 0], src[cb * 8 + 1]);
      p.y = pack2(src[cb * 8 + 2], src[cb * 8 + 3]);
      p.z = pack2(src[cb * 8 + 4], src[cb * 8 + 5]);
      p.w = pack2(src[cb * 8 + 6], src[cb * 8 + 7]);
      *(uint4*)&sQ[r * 64 + c32 + db * 8] = p;
    }
  }
  __syncthreads();
  bf16x16 aq0 = *(const bf16x16*)&sQ[mlo * 64 + half * 16];
  bf16x16 aq1 = *(const bf16x16*)&sQ[mlo * 64 + 32 + half * 16];

  f32x8 o[4];
#pragma unroll
  for (int j = 0; j < 4; ++j) o[j] = zero8();
  float mrow[8], lrow[8];
#pragma unroll
  for (int r = 0; r < 8; ++r) { mrow[r] = -3.0e30f; lrow[r] = 0.0f; }

  for (int kt0 = 0; kt0 < qt0 + 16; kt0 += 32) {
    __syncthreads();
    { // K tile: lane packs its key row (64 feats -> 8x b128 stores)
      const float* ks = base + (size_t)(kt0 + lane) * rs + D_;
#pragma unroll
      for (int cb = 0; cb < 8; ++cb) {
        uint4 p;
        p.x = pack2(ks[cb * 8 + 0], ks[cb * 8 + 1]);
        p.y = pack2(ks[cb * 8 + 2], ks[cb * 8 + 3]);
        p.z = pack2(ks[cb * 8 + 4], ks[cb * 8 + 5]);
        p.w = pack2(ks[cb * 8 + 6], ks[cb * 8 + 7]);
        *(uint4*)&sK[lane * 64 + cb * 8] = p;
      }
    }
    { // V tile transposed: lane owns dims d0,d0+1; pack 8 keys per b128 store
      int d0 = lane * 2;
      const float* vb = base + 2 * D_ + d0;
#pragma unroll
      for (int c8 = 0; c8 < 4; ++c8) {
        float va[8], vbv[8];
#pragma unroll
        for (int j = 0; j < 8; ++j) {
          const float* vp = vb + (size_t)(kt0 + c8 * 8 + j) * rs;
          va[j] = vp[0]; vbv[j] = vp[1];
        }
        uint4 pa, pb;
        pa.x = pack2(va[0], va[1]); pa.y = pack2(va[2], va[3]);
        pa.z = pack2(va[4], va[5]); pa.w = pack2(va[6], va[7]);
        pb.x = pack2(vbv[0], vbv[1]); pb.y = pack2(vbv[2], vbv[3]);
        pb.z = pack2(vbv[4], vbv[5]); pb.w = pack2(vbv[6], vbv[7]);
        *(uint4*)&sVt[d0 * 32 + c8 * 8] = pa;
        *(uint4*)&sVt[(d0 + 1) * 32 + c8 * 8] = pb;
      }
    }
    __syncthreads();
    // S = Q @ K^T over 32 keys -> two 16x16 c-frags (B-frags contiguous in sK rows)
    f32x8 s0 = zero8(), s1 = zero8();
    bf16x16 bk;
    bk = *(const bf16x16*)&sK[mlo * 64 + half * 16];
    s0 = __builtin_amdgcn_wmma_f32_16x16x32_bf16(false, aq0, false, bk, (short)0, s0, false, false);
    bk = *(const bf16x16*)&sK[mlo * 64 + 32 + half * 16];
    s0 = __builtin_amdgcn_wmma_f32_16x16x32_bf16(false, aq1, false, bk, (short)0, s0, false, false);
    bk = *(const bf16x16*)&sK[(16 + mlo) * 64 + half * 16];
    s1 = __builtin_amdgcn_wmma_f32_16x16x32_bf16(false, aq0, false, bk, (short)0, s1, false, false);
    bk = *(const bf16x16*)&sK[(16 + mlo) * 64 + 32 + half * 16];
    s1 = __builtin_amdgcn_wmma_f32_16x16x32_bf16(false, aq1, false, bk, (short)0, s1, false, false);

#pragma unroll
    for (int r = 0; r < 8; ++r) {
      int row = qt0 + half * 8 + r;
      float v0 = s0[r] * 0.125f;
      float v1 = s1[r] * 0.125f;
      if (kt0 + mlo > row)      v0 = -3.0e30f;
      if (kt0 + 16 + mlo > row) v1 = -3.0e30f;
      float mx = fmaxf(v0, v1);
#pragma unroll
      for (int off = 8; off >= 1; off >>= 1) mx = fmaxf(mx, __shfl_xor(mx, off, 32));
      float newm = fmaxf(mrow[r], mx);
      float fac  = __expf(mrow[r] - newm);
      float p0   = __expf(v0 - newm);
      float p1   = __expf(v1 - newm);
      float rsum = p0 + p1;
#pragma unroll
      for (int off = 8; off >= 1; off >>= 1) rsum += __shfl_xor(rsum, off, 32);
      lrow[r] = lrow[r] * fac + rsum;
      mrow[r] = newm;
      o[0][r] *= fac; o[1][r] *= fac; o[2][r] *= fac; o[3][r] *= fac;
      sP[(half * 8 + r) * 32 + mlo]      = (bf16)p0;
      sP[(half * 8 + r) * 32 + 16 + mlo] = (bf16)p1;
    }
    __syncthreads();
    bf16x16 pf = *(const bf16x16*)&sP[mlo * 32 + half * 16];
#pragma unroll
    for (int j = 0; j < 4; ++j) {
      bf16x16 vf = *(const bf16x16*)&sVt[(j * 16 + mlo) * 32 + half * 16];
      o[j] = __builtin_amdgcn_wmma_f32_16x16x32_bf16(false, pf, false, vf, (short)0, o[j], false, false);
    }
  }

#pragma unroll
  for (int r = 0; r < 8; ++r) {
    int row = qt0 + half * 8 + r;
    float inv = 1.0f / lrow[r];
#pragma unroll
    for (int j = 0; j < 4; ++j)
      out[(size_t)(b * T_ + row) * D_ + h * HD_ + j * 16 + mlo] = o[j][r] * inv;
  }
}

// ---- LayerNorm: one block per row ----
__global__ __launch_bounds__(256)
void layernorm_k(const float* __restrict__ x, const float* __restrict__ s,
                 const float* __restrict__ bsh, float* __restrict__ y)
{
  __shared__ float red[256];
  int row = blockIdx.x, tid = threadIdx.x;
  const float* xr = x + (size_t)row * D_;
  float sum = 0.f;
  for (int i = tid; i < D_; i += 256) sum += xr[i];
  red[tid] = sum; __syncthreads();
  for (int st = 128; st > 0; st >>= 1) { if (tid < st) red[tid] += red[tid + st]; __syncthreads(); }
  float mean = red[0] / (float)D_;
  __syncthreads();
  float vs = 0.f;
  for (int i = tid; i < D_; i += 256) { float d = xr[i] - mean; vs += d * d; }
  red[tid] = vs; __syncthreads();
  for (int st = 128; st > 0; st >>= 1) { if (tid < st) red[tid] += red[tid + st]; __syncthreads(); }
  float rstd = rsqrtf(red[0] / (float)D_ + 1e-5f);
  float* yr = y + (size_t)row * D_;
  for (int i = tid; i < D_; i += 256) yr[i] = (xr[i] - mean) * rstd * s[i] + bsh[i];
}

// ---- Embedding ----
__global__ void embed_k(const int* __restrict__ ids, const float* __restrict__ tok,
                        const float* __restrict__ pos, float* __restrict__ x)
{
  size_t i = (size_t)blockIdx.x * blockDim.x + threadIdx.x;
  size_t total = (size_t)B_ * T_ * D_;
  if (i >= total) return;
  int d  = (int)(i % D_);
  int bt = (int)(i / D_);
  int t  = bt % T_;
  int id = ids[bt];
  x[i] = tok[(size_t)id * D_ + d] + pos[(size_t)t * D_ + d];
}

// ---- f32 -> bf16 A-operand convert with swz32 k-permutation ----
__global__ void cvt_a_swz_k(const float* __restrict__ src, bf16* __restrict__ dst,
                            size_t n, int K)
{
  size_t i = (size_t)blockIdx.x * blockDim.x + threadIdx.x;
  size_t stride = (size_t)gridDim.x * blockDim.x;
  for (; i < n; i += stride) {
    size_t m = i / (size_t)K;
    int kk = (int)(i - m * (size_t)K);
    int kb = kk & ~31;
    int o  = kk & 31;
    dst[i] = (bf16)src[m * (size_t)K + kb + swz32(o)];
  }
}

// ---- f32 -> bf16 B-operand convert into tiled column-major layout ----
// dst tile (kt,nt): ((nt*Kt + kt)*64 + n)*32 + k ; zero-padded for n >= N.
__global__ __launch_bounds__(256)
void cvt_b_swz_k(const float* __restrict__ src, bf16* __restrict__ dst, int K, int N)
{
  __shared__ float tile[32 * 64];
  const int kt = blockIdx.x, nt = blockIdx.y, t = threadIdx.x;
  {
    int row = t >> 3, c0 = (t & 7) * 8;
    const float* sp = src + (size_t)(kt * 32 + row) * N + nt * 64 + c0;
#pragma unroll
    for (int j = 0; j < 8; ++j) {
      int n = nt * 64 + c0 + j;
      tile[row * 64 + c0 + j] = (n < N) ? sp[j] : 0.f;
    }
  }
  __syncthreads();
  {
    int n = t >> 2, k0 = (t & 3) * 8;
    bf16* dp = dst + (((size_t)nt * gridDim.x + kt) * 64 + n) * 32 + k0;
#pragma unroll
    for (int j = 0; j < 8; ++j) dp[j] = (bf16)tile[swz32(k0 + j) * 64 + n];
  }
}

// ---- Memory module: state init + sequential scan ----
__global__ void mem_init_k(const float* __restrict__ M1_0, const float* __restrict__ M2_0,
                           float* __restrict__ state)
{
  int b = blockIdx.x;
  const int SZ = MLP_ * D_;
  float* st = state + (size_t)b * 4 * SZ;
  for (int i = threadIdx.x; i < SZ; i += blockDim.x) {
    st[i]          = M1_0[i];
    st[SZ + i]     = M2_0[i];
    st[2 * SZ + i] = 0.f;
    st[3 * SZ + i] = 0.f;
  }
}

__global__ __launch_bounds__(768)
void mem_scan_k(const float* __restrict__ kin, const float* __restrict__ qin,
                const float* __restrict__ vin, float* __restrict__ x,
                float* __restrict__ state,
                const float* __restrict__ gw, const float* __restrict__ gu,
                const float* __restrict__ gb)
{
  __shared__ float skt[D_], sqt[D_], svt[D_], sxt[D_], serr[D_], sred[D_];
  __shared__ float sgp[MLP_ * 12];
  __shared__ float sa[MLP_], sda[MLP_], saq[MLP_];
  __shared__ float s_surprise, s_gate;
  const int tid = threadIdx.x;
  const int b   = blockIdx.x;
  const int SZ  = MLP_ * D_;
  float* M1 = state + (size_t)b * 4 * SZ;
  float* M2 = M1 + SZ;
  float* S1 = M2 + SZ;
  float* S2 = S1 + SZ;
  const int g = tid / 12, lg = tid % 12;  // 64 groups x 12 threads

  for (int t = 0; t < T_; ++t) {
    size_t roff = ((size_t)b * T_ + t) * D_;
    skt[tid] = kin[roff + tid];
    sqt[tid] = qin[roff + tid];
    svt[tid] = vin[roff + tid];
    sxt[tid] = x[roff + tid];
    __syncthreads();
    { // a = tanh(M1 @ k)
      const float* r1 = M1 + (size_t)g * D_ + lg * 64;
      const float* kk = skt + lg * 64;
      float p = 0.f;
      for (int j = 0; j < 64; ++j) p += r1[j] * kk[j];
      sgp[g * 12 + lg] = p;
    }
    __syncthreads();
    if (tid < MLP_) {
      float sum = 0.f;
      for (int j = 0; j < 12; ++j) sum += sgp[tid * 12 + j];
      sa[tid] = tanhf(sum);
    }
    __syncthreads();
    { // err = M2 @ a - v ; surprise partials
      const float* r2 = M2 + (size_t)tid * MLP_;
      float e = -svt[tid];
      for (int j = 0; j < MLP_; ++j) e += r2[j] * sa[j];
      serr[tid] = e;
      sred[tid] = e * e;
    }
    __syncthreads();
    for (int st = 512; st > 0; st >>= 1) {
      if (tid < st && tid + st < D_) sred[tid] += sred[tid + st];
      __syncthreads();
    }
    if (tid == 0) s_surprise = sqrtf(sred[0]);
    { // da (uses OLD M2)
      float p = 0.f;
      for (int j = 0; j < 64; ++j) {
        int d = lg * 64 + j;
        p += M2[(size_t)d * MLP_ + g] * serr[d];
      }
      sgp[g * 12 + lg] = p;
    }
    __syncthreads();
    if (tid < MLP_) {
      float sum = 0.f;
      for (int j = 0; j < 12; ++j) sum += sgp[tid * 12 + j];
      float a = sa[tid];
      sda[tid] = sum * (1.f - a * a);
    }
    __syncthreads();
    // momentum/decay updates
    for (int i = tid; i < SZ; i += D_) {
      int d = i >> 6, h = i & 63;
      float g2 = serr[d] * sa[h];
      float s2 = 0.9f * S2[i] - 0.1f * g2;
      S2[i] = s2;
      M2[i] = 0.99f * M2[i] + s2;
    }
    for (int i = tid; i < SZ; i += D_) {
      int hh = i / D_, d = i % D_;
      float g1 = sda[hh] * skt[d];
      float s1 = 0.9f * S1[i] - 0.1f * g1;
      S1[i] = s1;
      M1[i] = 0.99f * M1[i] + s1;
    }
    __syncthreads();
    { // aq = tanh(M1n @ q)
      const float* r1 = M1 + (size_t)g * D_ + lg * 64;
      const float* qq = sqt + lg * 64;
      float p = 0.f;
      for (int j = 0; j < 64; ++j) p += r1[j] * qq[j];
      sgp[g * 12 + lg] = p;
    }
    __syncthreads();
    if (tid < MLP_) {
      float sum = 0.f;
      for (int j = 0; j < 12; ++j) sum += sgp[tid * 12 + j];
      saq[tid] = tanhf(sum);
    }
    sred[tid] = sxt[tid] * gw[tid];
    __syncthreads();
    for (int st = 512; st > 0; st >>= 1) {
      if (tid < st && tid + st < D_) sred[tid] += sred[tid + st];
      __syncthreads();
    }
    if (tid == 0) {
      float z = sred[0] + s_surprise * gu[0] + gb[0];
      s_gate = 1.f / (1.f + __expf(-z));
    }
    __syncthreads();
    { // r = M2n @ aq ; x += gate * r
      const float* r2 = M2 + (size_t)tid * MLP_;
      float rr = 0.f;
      for (int j = 0; j < MLP_; ++j) rr += r2[j] * saq[j];
      x[roff + tid] += s_gate * rr;
    }
    __syncthreads();
  }
}

// ---- Host orchestration ----
extern "C" void kernel_launch(void* const* d_in, const int* in_sizes, int n_in,
                              void* d_out, int out_size, void* d_ws, size_t ws_size,
                              hipStream_t stream)
{
  (void)in_sizes; (void)n_in; (void)out_size; (void)ws_size;
  const int*   ids   = (const int*)  d_in[0];
  const float* tok   = (const float*)d_in[1];
  const float* pos   = (const float*)d_in[2];
  const float* ln1s  = (const float*)d_in[3];
  const float* ln1b  = (const float*)d_in[4];
  const float* Wqkv  = (const float*)d_in[5];
  const float* Wo    = (const float*)d_in[6];
  const float* ln2s  = (const float*)d_in[7];
  const float* ln2b  = (const float*)d_in[8];
  const float* Wf1   = (const float*)d_in[9];
  const float* bf1   = (const float*)d_in[10];
  const float* Wf2   = (const float*)d_in[11];
  const float* bf2   = (const float*)d_in[12];
  const float* lnfs  = (const float*)d_in[13];
  const float* lnfb  = (const float*)d_in[14];
  const float* headW = (const float*)d_in[15];
  const float* Wk    = (const float*)d_in[16];
  const float* Wq    = (const float*)d_in[17];
  const float* Wv    = (const float*)d_in[18];
  const float* M1_0  = (const float*)d_in[19];
  const float* M2_0  = (const float*)d_in[20];
  const float* gatew = (const float*)d_in[21];
  const float* gateu = (const float*)d_in[22];
  const float* gateb = (const float*)d_in[23];

  char* wsp = (char*)d_ws;
  size_t off = 0;
  auto alloc = [&](size_t bytes) {
    void* p = wsp + off;
    off += (bytes + 255) & ~(size_t)255;
    return p;
  };
  const size_t BT = (size_t)B_ * T_;
  const size_t Vpad = (size_t)((V_ + 63) / 64) * 64;
  float* x      = (float*)alloc(BT * D_ * 4);
  float* lnout  = (float*)alloc(BT * D_ * 4);
  float* big    = (float*)alloc(BT * DFF_ * 4);
  float* attno  = (float*)alloc(BT * D_ * 4);
  bf16*  abuf   = (bf16*) alloc(BT * DFF_ * 2);
  bf16*  wbuf   = (bf16*) alloc((size_t)D_ * Vpad * 2);
  float* mstate = (float*)alloc((size_t)B_ * 4 * MLP_ * D_ * 4);

  auto cvt_a = [&](const float* s, bf16* d, size_t n, int K) {
    unsigned blocks = (unsigned)((n + 255) / 256);
    if (blocks > 16384) blocks = 16384;
    cvt_a_swz_k<<<blocks, 256, 0, stream>>>(s, d, n, K);
  };
  auto cvt_b = [&](const float* s, bf16* d, int K, int N) {
    dim3 grid(K / 32, (N + 63) / 64);
    cvt_b_swz_k<<<grid, 256, 0, stream>>>(s, d, K, N);
  };
  auto gemm = [&](const bf16* A, const bf16* Bm, const float* bias, const float* res,
                  float* C, int M, int N, int K, int flags) {
    dim3 grid((N + 63) / 64, M / 64);
    switch (flags) {
      case 0:
        gemm_bf16<0><<<grid, 128, 0, stream>>>(A, Bm, bias, res, C, M, N, K, K / 32);
        break;
      case GF_RES:
        gemm_bf16<GF_RES><<<grid, 128, 0, stream>>>(A, Bm, bias, res, C, M, N, K, K / 32);
        break;
      case GF_BIAS | GF_GELU:
        gemm_bf16<GF_BIAS | GF_GELU><<<grid, 128, 0, stream>>>(A, Bm, bias, res, C, M, N, K, K / 32);
        break;
      default:
        gemm_bf16<GF_BIAS | GF_RES><<<grid, 128, 0, stream>>>(A, Bm, bias, res, C, M, N, K, K / 32);
        break;
    }
  };

  embed_k<<<(unsigned)((BT * D_ + 255) / 256), 256, 0, stream>>>(ids, tok, pos, x);

  for (int i = 0; i < NL_; ++i) {
    if (i == 0 || i == 2) {
      int m = i / 2;
      cvt_a(x, abuf, BT * D_, D_);
      cvt_b(Wk + (size_t)m * D_ * D_, wbuf, D_, D_);
      gemm(abuf, wbuf, nullptr, nullptr, big + 0 * BT * D_, (int)BT, D_, D_, 0);
      cvt_b(Wq + (size_t)m * D_ * D_, wbuf, D_, D_);
      gemm(abuf, wbuf, nullptr, nullptr, big + 1 * BT * D_, (int)BT, D_, D_, 0);
      cvt_b(Wv + (size_t)m * D_ * D_, wbuf, D_, D_);
      gemm(abuf, wbuf, nullptr, nullptr, big + 2 * BT * D_, (int)BT, D_, D_, 0);
      mem_init_k<<<B_, 256, 0, stream>>>(M1_0 + (size_t)m * MLP_ * D_,
                                         M2_0 + (size_t)m * D_ * MLP_, mstate);
      mem_scan_k<<<B_, D_, 0, stream>>>(big, big + BT * D_, big + 2 * BT * D_, x, mstate,
                                        gatew + (size_t)m * D_, gateu + m, gateb + m);
    }
    // attention block
    layernorm_k<<<(unsigned)BT, 256, 0, stream>>>(x, ln1s + (size_t)i * D_, ln1b + (size_t)i * D_, lnout);
    cvt_a(lnout, abuf, BT * D_, D_);
    cvt_b(Wqkv + (size_t)i * D_ * 3 * D_, wbuf, D_, 3 * D_);
    gemm(abuf, wbuf, nullptr, nullptr, big, (int)BT, 3 * D_, D_, 0);
    attn_fa<<<dim3(T_ / 16, H_, B_), 32, 0, stream>>>(big, attno);
    cvt_a(attno, abuf, BT * D_, D_);
    cvt_b(Wo + (size_t)i * D_ * D_, wbuf, D_, D_);
    gemm(abuf, wbuf, nullptr, x, x, (int)BT, D_, D_, GF_RES);
    // MLP block
    layernorm_k<<<(unsigned)BT, 256, 0, stream>>>(x, ln2s + (size_t)i * D_, ln2b + (size_t)i * D_, lnout);
    cvt_a(lnout, abuf, BT * D_, D_);
    cvt_b(Wf1 + (size_t)i * D_ * DFF_, wbuf, D_, DFF_);
    gemm(abuf, wbuf, bf1 + (size_t)i * DFF_, nullptr, big, (int)BT, DFF_, D_, GF_BIAS | GF_GELU);
    cvt_a(big, abuf, BT * DFF_, DFF_);
    cvt_b(Wf2 + (size_t)i * DFF_ * D_, wbuf, DFF_, D_);
    gemm(abuf, wbuf, bf2 + (size_t)i * D_, x, x, (int)BT, D_, DFF_, GF_BIAS | GF_RES);
  }

  layernorm_k<<<(unsigned)BT, 256, 0, stream>>>(x, lnfs, lnfb, lnout);
  cvt_a(lnout, abuf, BT * D_, D_);
  cvt_b(headW, wbuf, D_, V_);
  gemm(abuf, wbuf, nullptr, nullptr, (float*)d_out, (int)BT, V_, D_, 0);
}
